// PitchExtractor_15006615733686
// MI455X (gfx1250) — compile-verified, matrix operand order checked by reading, not run on hardware
//
#include <hip/hip_runtime.h>
#include <math.h>

typedef _Float16 half_t;
typedef __attribute__((ext_vector_type(16))) _Float16 v16h;
typedef __attribute__((ext_vector_type(8)))  _Float16 v8h;
typedef __attribute__((ext_vector_type(8)))  float    v8f;

constexpr int WIN   = 2048;
constexpr int HOP   = 256;
constexpr int K     = WIN;         // GEMM reduction dim
constexpr int F     = WIN / 2 + 1; // 1025 rfft bins
constexpr int NPAD  = 1040;        // F padded to multiple of 16
constexpr int NTILES = NPAD / 16;  // 65
constexpr float TWO_PI = 6.283185307179586f;

// ---------------------------------------------------------------------------
// 1) Build windowed frame matrix A[Mpad][K] (f16), reflect-padded audio,
//    periodic Hann window. Rows >= T are zero.
// ---------------------------------------------------------------------------
__global__ void build_frames_kernel(const float* __restrict__ audio,
                                    half_t* __restrict__ A,
                                    int n_audio, int T, long total)
{
  long idx = (long)blockIdx.x * blockDim.x + threadIdx.x;
  if (idx >= total) return;
  int k = (int)(idx & (K - 1));
  int m = (int)(idx >> 11);          // K == 2048
  float v = 0.0f;
  if (m < T) {
    long j = (long)m * HOP + k - (WIN / 2);
    if (j < 0) j = -j;                               // reflect left
    if (j >= n_audio) j = 2L * n_audio - 2 - j;      // reflect right
    float w = 0.5f * (1.0f - __cosf((float)k * (TWO_PI / (float)WIN)));
    v = audio[j] * w;
  }
  A[idx] = (half_t)v;
}

// ---------------------------------------------------------------------------
// 2) DFT basis stored N-major (transposed) so B-fragment loads are two
//    contiguous b128s per lane. Exact integer phase reduction mod 2048.
// ---------------------------------------------------------------------------
__global__ void build_basis_kernel(half_t* __restrict__ Bc,
                                   half_t* __restrict__ Bs, long total)
{
  long idx = (long)blockIdx.x * blockDim.x + threadIdx.x;
  if (idx >= total) return;
  int k = (int)(idx & (K - 1));
  int n = (int)(idx >> 11);
  int p = (n * k) & (K - 1);
  float ang = (float)p * (TWO_PI / (float)WIN);
  Bc[idx] = (half_t)__cosf(ang);
  Bs[idx] = (half_t)__sinf(ang);
}

// ---------------------------------------------------------------------------
// 3) STFT as WMMA GEMM. One wave -> 32x16 output block (2 M-tiles x 1 N-tile)
//    so each Bc/Bs fragment feeds two A fragments: 8 b128 loads per 4 WMMAs.
//    Register double-buffering: next K-step's loads issue before the current
//    step's 4 WMMAs, so v_wmma executes under outstanding loadcnt.
//    Fragment layout per ISA 7.12.2 (16-bit, 16x32): lane L -> row r = L&15,
//    half h = L>>4; elems[0..7] = K0+h*8.., elems[8..15] = K0+16+h*8..
// ---------------------------------------------------------------------------
__global__ __launch_bounds__(32) void stft_wmma_kernel(
    const half_t* __restrict__ A, const half_t* __restrict__ Bc,
    const half_t* __restrict__ Bs, float* __restrict__ mag)
{
  const int nT   = blockIdx.x;     // freq tile
  const int mB   = blockIdx.y;     // 32-row frame block
  const int lane = threadIdx.x & 31;
  const int r    = lane & 15;
  const int h    = lane >> 4;

  const half_t* A0p = A  + (size_t)(mB * 32 + r) * K + h * 8;
  const half_t* A1p = A0p + (size_t)16 * K;
  const half_t* Bcp = Bc + (size_t)(nT * 16 + r) * K + h * 8;
  const half_t* Bsp = Bs + (size_t)(nT * 16 + r) * K + h * 8;

  v8f acc0r = {}, acc0i = {}, acc1r = {}, acc1i = {};

  // prologue: fragments for k0 = 0
  v8h a0lo = *(const v8h*)(A0p);      v8h a0hi = *(const v8h*)(A0p + 16);
  v8h a1lo = *(const v8h*)(A1p);      v8h a1hi = *(const v8h*)(A1p + 16);
  v8h bclo = *(const v8h*)(Bcp);      v8h bchi = *(const v8h*)(Bcp + 16);
  v8h bslo = *(const v8h*)(Bsp);      v8h bshi = *(const v8h*)(Bsp + 16);

  for (int kn = 32; kn < K; kn += 32) {
    // issue next step's loads first (in flight during this step's WMMAs)
    v8h na0lo = *(const v8h*)(A0p + kn);      v8h na0hi = *(const v8h*)(A0p + kn + 16);
    v8h na1lo = *(const v8h*)(A1p + kn);      v8h na1hi = *(const v8h*)(A1p + kn + 16);
    v8h nbclo = *(const v8h*)(Bcp + kn);      v8h nbchi = *(const v8h*)(Bcp + kn + 16);
    v8h nbslo = *(const v8h*)(Bsp + kn);      v8h nbshi = *(const v8h*)(Bsp + kn + 16);

    __builtin_prefetch(A0p + kn + 512, 0, 1);   // global_prefetch_b8 (speculative)
    __builtin_prefetch(A1p + kn + 512, 0, 1);

    v16h a0, a1, bcv, bsv;
#pragma unroll
    for (int i = 0; i < 8; ++i) {
      a0[i]  = a0lo[i];  a0[i + 8]  = a0hi[i];
      a1[i]  = a1lo[i];  a1[i + 8]  = a1hi[i];
      bcv[i] = bclo[i];  bcv[i + 8] = bchi[i];
      bsv[i] = bslo[i];  bsv[i + 8] = bshi[i];
    }
    acc0r = __builtin_amdgcn_wmma_f32_16x16x32_f16(false, a0, false, bcv, (short)0, acc0r, false, false);
    acc0i = __builtin_amdgcn_wmma_f32_16x16x32_f16(false, a0, false, bsv, (short)0, acc0i, false, false);
    acc1r = __builtin_amdgcn_wmma_f32_16x16x32_f16(false, a1, false, bcv, (short)0, acc1r, false, false);
    acc1i = __builtin_amdgcn_wmma_f32_16x16x32_f16(false, a1, false, bsv, (short)0, acc1i, false, false);

    a0lo = na0lo; a0hi = na0hi; a1lo = na1lo; a1hi = na1hi;
    bclo = nbclo; bchi = nbchi; bslo = nbslo; bshi = nbshi;
  }

  // epilogue: last K step
  {
    v16h a0, a1, bcv, bsv;
#pragma unroll
    for (int i = 0; i < 8; ++i) {
      a0[i]  = a0lo[i];  a0[i + 8]  = a0hi[i];
      a1[i]  = a1lo[i];  a1[i + 8]  = a1hi[i];
      bcv[i] = bclo[i];  bcv[i + 8] = bchi[i];
      bsv[i] = bslo[i];  bsv[i + 8] = bshi[i];
    }
    acc0r = __builtin_amdgcn_wmma_f32_16x16x32_f16(false, a0, false, bcv, (short)0, acc0r, false, false);
    acc0i = __builtin_amdgcn_wmma_f32_16x16x32_f16(false, a0, false, bsv, (short)0, acc0i, false, false);
    acc1r = __builtin_amdgcn_wmma_f32_16x16x32_f16(false, a1, false, bcv, (short)0, acc1r, false, false);
    acc1i = __builtin_amdgcn_wmma_f32_16x16x32_f16(false, a1, false, bsv, (short)0, acc1i, false, false);
  }

  // D layout: VGPR j -> row m = h*8 + j, col n = lane&15. Fused |X| epilogue.
#pragma unroll
  for (int j = 0; j < 8; ++j) {
    int m = h * 8 + j;
    size_t row = (size_t)(mB * 32 + m);
    float r0 = acc0r[j], i0 = acc0i[j];
    float r1 = acc1r[j], i1 = acc1i[j];
    mag[row * NPAD + nT * 16 + r]        = __builtin_sqrtf(r0 * r0 + i0 * i0);
    mag[(row + 16) * NPAD + nT * 16 + r] = __builtin_sqrtf(r1 * r1 + i1 * i1);
  }
}

// ---------------------------------------------------------------------------
// 4) Per-frame spectral autocorrelation (lags 1..511) from LDS-cached row,
//    then "any interior local max in ac[1:]" test (ac zero for lag>=512,
//    matching the reference's valid-lag mask).
// ---------------------------------------------------------------------------
__global__ __launch_bounds__(512) void autocorr_peak_kernel(
    const float* __restrict__ mag, float* __restrict__ f0)
{
  __shared__ float sm[1056];
  __shared__ float ac[1056];
  __shared__ int   flag;
  const int t   = blockIdx.x;
  const int tid = threadIdx.x;
  const float* row = mag + (size_t)t * NPAD;

  for (int i = tid; i < 1056; i += 512) {
    sm[i] = (i < F) ? row[i] : 0.0f;
    ac[i] = 0.0f;
  }
  if (tid == 0) flag = 0;
  __syncthreads();

  if (tid >= 1 && tid < F / 2) {            // lags 1..511
    const int l = tid;
    float s = 0.0f;
    for (int k = 0; k + l < F; ++k) s = fmaf(sm[k], sm[k + l], s);
    ac[l] = s;
  }
  __syncthreads();

  bool p = false;
  for (int l = tid; l < 1024; l += 512) {   // interior positions of ac[1:]
    if (l >= 2 && ac[l] > ac[l - 1] && ac[l] > ac[l + 1]) p = true;
  }
  if (p) atomicOr(&flag, 1);
  __syncthreads();

  if (tid == 0) f0[t] = flag ? 50.0f : 0.0f;
}

// ---------------------------------------------------------------------------
// 5) Median-of-5 with edge padding (9-comparator sorting network).
// ---------------------------------------------------------------------------
__global__ void median5_kernel(const float* __restrict__ f0,
                               float* __restrict__ out, int T)
{
  int t = blockIdx.x * blockDim.x + threadIdx.x;
  if (t >= T) return;
  float v[5];
#pragma unroll
  for (int i = 0; i < 5; ++i) {
    int j = t + i - 2;
    j = j < 0 ? 0 : (j > T - 1 ? T - 1 : j);
    v[i] = f0[j];
  }
#define CSWAP(a, b) { float lo = fminf(v[a], v[b]); v[b] = fmaxf(v[a], v[b]); v[a] = lo; }
  CSWAP(0, 1) CSWAP(3, 4) CSWAP(2, 4) CSWAP(2, 3) CSWAP(0, 3)
  CSWAP(0, 2) CSWAP(1, 4) CSWAP(1, 3) CSWAP(1, 2)
#undef CSWAP
  out[t] = v[2];
}

// ---------------------------------------------------------------------------
extern "C" void kernel_launch(void* const* d_in, const int* in_sizes, int n_in,
                              void* d_out, int out_size, void* d_ws, size_t ws_size,
                              hipStream_t stream)
{
  const float* audio = (const float*)d_in[0];
  const int n_audio  = in_sizes[0];             // 2,097,152
  const int T        = n_audio / HOP + 1;       // 8193
  const int mBlocks  = (T + 31) / 32;           // 257 (32-row GEMM blocks)
  const int Mpad     = mBlocks * 32;            // 8224
  float* out = (float*)d_out;
  (void)n_in; (void)out_size; (void)ws_size;

  char* ws = (char*)d_ws;
  size_t off = 0;
  auto wsalloc = [&](size_t bytes) -> void* {
    void* p = ws + off;
    off = (off + bytes + 255) & ~(size_t)255;
    return p;
  };
  half_t* A   = (half_t*)wsalloc((size_t)Mpad * K * sizeof(half_t));   // 33.7 MB
  half_t* Bc  = (half_t*)wsalloc((size_t)NPAD * K * sizeof(half_t));   //  4.3 MB
  half_t* Bs  = (half_t*)wsalloc((size_t)NPAD * K * sizeof(half_t));   //  4.3 MB
  float*  mag = (float*)wsalloc((size_t)Mpad * NPAD * sizeof(float));  // 34.2 MB
  float*  f0  = (float*)wsalloc((size_t)Mpad * sizeof(float));

  long totA = (long)Mpad * K;
  build_frames_kernel<<<(int)((totA + 255) / 256), 256, 0, stream>>>(
      audio, A, n_audio, T, totA);

  long totB = (long)NPAD * K;
  build_basis_kernel<<<(int)((totB + 255) / 256), 256, 0, stream>>>(Bc, Bs, totB);

  stft_wmma_kernel<<<dim3(NTILES, mBlocks), 32, 0, stream>>>(A, Bc, Bs, mag);

  autocorr_peak_kernel<<<T, 512, 0, stream>>>(mag, f0);

  median5_kernel<<<(T + 255) / 256, 256, 0, stream>>>(f0, out, T);
}
